// Non_Local_Sparse_Network_83837761618154
// MI455X (gfx1250) — matrix operand order repaired and checked by reading, not compiled, not run on hardware
//
#include <hip/hip_runtime.h>
#include <hip/hip_bf16.h>
#include <math.h>

typedef __attribute__((ext_vector_type(16))) __bf16 v16bf;
typedef __attribute__((ext_vector_type(8)))  float  v8f;

// ---------------- problem constants ----------------
#define Bn     2
#define C0n    64
#define Cmn    16
#define HWn    100
#define Ln     10000          // H*W
#define L2n    20000          // 2*L tokens
#define NHASH  4
#define UTOT   80000          // NHASH * 2L
#define NCHK   139            // ceil(20000/144)
#define CHK    144
#define NBK    512            // hash buckets = 4*128
#define NBLK   79             // sort blocks of 1024 per batch
#define WSTR   448            // padded K for gemm2 (432 -> 448)
#define SMEM_ATT 226112

static __device__ inline unsigned pcg(unsigned x) {
  x = x * 747796405u + 2891336453u;
  unsigned w = ((x >> ((x >> 28u) + 4u)) ^ x) * 277803737u;
  return (w >> 22u) ^ w;
}

// ---------------- rot (deterministic stand-in for jax normal) ----------------
__global__ void rot_kernel(float* __restrict__ rot) {
  int id = blockIdx.x * 256 + threadIdx.x;
  if (id >= 16 * NHASH * 64) return;
  unsigned a = pcg((unsigned)id * 2654435769u ^ 42u);
  unsigned b = pcg(a ^ 0x9e3779b9u);
  float u1 = ((float)a + 1.0f) * 2.3283064365386963e-10f;   // (0,1]
  float u2 = (float)b * 2.3283064365386963e-10f;
  rot[id] = sqrtf(-2.0f * logf(u1)) * cosf(6.28318530718f * u2);
}

// ---------------- generic 3x3 conv + bias (+optional relu), NCHW ----------------
__global__ void conv3x3_kernel(const float* __restrict__ in, const float* __restrict__ w,
                               const float* __restrict__ bias, float* __restrict__ out,
                               int Cin, int Cout, int relu) {
  int id = blockIdx.x * blockDim.x + threadIdx.x;
  int total = Bn * Cout * HWn * HWn;
  if (id >= total) return;
  int x = id % HWn, y = (id / HWn) % HWn;
  int co = (id / (HWn * HWn)) % Cout;
  int b  = id / (HWn * HWn * Cout);
  float acc = bias[co];
  for (int ci = 0; ci < Cin; ++ci) {
    const float* ip = in + ((size_t)(b * Cin + ci) * HWn) * HWn;
    const float* wp = w + ((size_t)co * Cin + ci) * 9;
    for (int ky = 0; ky < 3; ++ky) {
      int yy = y + ky - 1; if (yy < 0 || yy >= HWn) continue;
      for (int kx = 0; kx < 3; ++kx) {
        int xx = x + kx - 1; if (xx < 0 || xx >= HWn) continue;
        acc += ip[yy * HWn + xx] * wp[ky * 3 + kx];
      }
    }
  }
  if (relu) acc = fmaxf(acc, 0.0f);
  out[id] = acc;
}

// ---------------- m-branch stage2: conv3x3(tmp16) + 1x1 skip -> f1f2 rows ----------------
__global__ void m_stage2_kernel(const float* __restrict__ fd, const float* __restrict__ tmp16,
                                const float* __restrict__ w2, const float* __restrict__ b2,
                                const float* __restrict__ wsk, const float* __restrict__ bsk,
                                float* __restrict__ f1f2, int toff) {
  int id = blockIdx.x * blockDim.x + threadIdx.x;
  int total = Bn * Cmn * HWn * HWn;
  if (id >= total) return;
  int x = id % HWn, y = (id / HWn) % HWn;
  int co = (id / (HWn * HWn)) % Cmn;
  int b  = id / (HWn * HWn * Cmn);
  float acc = b2[co];
  for (int ci = 0; ci < Cmn; ++ci) {
    const float* ip = tmp16 + ((size_t)(b * Cmn + ci) * HWn) * HWn;
    const float* wp = w2 + ((size_t)co * Cmn + ci) * 9;
    for (int ky = 0; ky < 3; ++ky) {
      int yy = y + ky - 1; if (yy < 0 || yy >= HWn) continue;
      for (int kx = 0; kx < 3; ++kx) {
        int xx = x + kx - 1; if (xx < 0 || xx >= HWn) continue;
        acc += ip[yy * HWn + xx] * wp[ky * 3 + kx];
      }
    }
  }
  float sk = bsk[co];
  for (int ci = 0; ci < C0n; ++ci)
    sk += fd[((size_t)(b * C0n + ci) * HWn + y) * HWn + x] * wsk[co * C0n + ci];
  int t = y * HWn + x;
  f1f2[((size_t)b * L2n + toff + t) * Cmn + co] = sk + acc;
}

// ---------------- aux stage2: conv3x3(tmp64) + identity skip -> refw rows ----------------
__global__ void aux_stage2_kernel(const float* __restrict__ fd, const float* __restrict__ tmp64,
                                  const float* __restrict__ w2, const float* __restrict__ b2,
                                  float* __restrict__ refw, int toff) {
  int id = blockIdx.x * blockDim.x + threadIdx.x;
  int total = Bn * C0n * HWn * HWn;
  if (id >= total) return;
  int x = id % HWn, y = (id / HWn) % HWn;
  int co = (id / (HWn * HWn)) % C0n;
  int b  = id / (HWn * HWn * C0n);
  float acc = b2[co];
  for (int ci = 0; ci < C0n; ++ci) {
    const float* ip = tmp64 + ((size_t)(b * C0n + ci) * HWn) * HWn;
    const float* wp = w2 + ((size_t)co * C0n + ci) * 9;
    for (int ky = 0; ky < 3; ++ky) {
      int yy = y + ky - 1; if (yy < 0 || yy >= HWn) continue;
      for (int kx = 0; kx < 3; ++kx) {
        int xx = x + kx - 1; if (xx < 0 || xx >= HWn) continue;
        acc += ip[yy * HWn + xx] * wp[ky * 3 + kx];
      }
    }
  }
  acc += fd[((size_t)(b * C0n + co) * HWn + y) * HWn + x];
  int t = y * HWn + x;
  refw[((size_t)b * L2n + toff + t) * C0n + co] = acc;
}

// ---------------- LSH codes (permuted layout) ----------------
__global__ void codes_kernel(const float* __restrict__ f1f2, const float* __restrict__ rot,
                             const int* __restrict__ rindex, int* __restrict__ ckey) {
  int id = blockIdx.x * blockDim.x + threadIdx.x;
  if (id >= Bn * UTOT) return;
  int tp = id % L2n;
  int h  = (id / L2n) % NHASH;
  int b  = id / UTOT;
  int t  = rindex[tp];
  const float* f = f1f2 + ((size_t)b * L2n + t) * Cmn;
  float fv[16];
#pragma unroll
  for (int e = 0; e < 16; ++e) fv[e] = f[e];
  float best = -INFINITY; int bi = 0;
  for (int i = 0; i < 64; ++i) {          // positives first (argmax first-occurrence)
    float s = 0.0f;
#pragma unroll
    for (int e = 0; e < 16; ++e) s += fv[e] * rot[(e * NHASH + h) * 64 + i];
    if (s > best) { best = s; bi = i; }
  }
  for (int i = 0; i < 64; ++i) {
    float s = 0.0f;
#pragma unroll
    for (int e = 0; e < 16; ++e) s += fv[e] * rot[(e * NHASH + h) * 64 + i];
    s = -s;
    if (s > best) { best = s; bi = i + 64; }
  }
  ckey[(size_t)b * UTOT + h * L2n + tp] = bi + h * 128;
}

// ---------------- stable 512-bucket counting sort (argsort) ----------------
__global__ void hist_kernel(const int* __restrict__ ckey, int* __restrict__ hist) {
  __shared__ int hl[NBK];
  int blk = blockIdx.x, b = blk / NBLK, j = blk % NBLK;
  for (int v = threadIdx.x; v < NBK; v += 256) hl[v] = 0;
  __syncthreads();
  for (int q = 0; q < 4; ++q) {
    int u = j * 1024 + q * 256 + threadIdx.x;
    if (u < UTOT) atomicAdd(&hl[ckey[(size_t)b * UTOT + u]], 1);
  }
  __syncthreads();
  for (int v = threadIdx.x; v < NBK; v += 256) hist[((size_t)b * NBLK + j) * NBK + v] = hl[v];
}

__global__ void scan_kernel(const int* __restrict__ hist, int* __restrict__ offs) {
  __shared__ int sc[NBK];
  int b = blockIdx.x, v = threadIdx.x;
  int tot = 0;
  for (int j = 0; j < NBLK; ++j) tot += hist[((size_t)b * NBLK + j) * NBK + v];
  sc[v] = tot;
  __syncthreads();
  for (int off = 1; off < NBK; off <<= 1) {
    int val = (v >= off) ? sc[v - off] : 0;
    __syncthreads();
    sc[v] += val;
    __syncthreads();
  }
  int run = sc[v] - tot;                    // exclusive base across buckets
  for (int j = 0; j < NBLK; ++j) {
    offs[((size_t)b * NBLK + j) * NBK + v] = run;
    run += hist[((size_t)b * NBLK + j) * NBK + v];
  }
}

__global__ void scatter_kernel(const int* __restrict__ ckey, const int* __restrict__ offs,
                               int* __restrict__ sidx) {
  __shared__ int cnt[NBK];
  __shared__ int kb[256];
  int blk = blockIdx.x, b = blk / NBLK, j = blk % NBLK;
  for (int v = threadIdx.x; v < NBK; v += 256) cnt[v] = 0;
  __syncthreads();
  for (int q = 0; q < 4; ++q) {
    int u = j * 1024 + q * 256 + threadIdx.x;
    int key = (u < UTOT) ? ckey[(size_t)b * UTOT + u] : -1;
    kb[threadIdx.x] = key;
    __syncthreads();
    if (key >= 0) {
      int rank = 0;
      for (int p = 0; p < (int)threadIdx.x; ++p) rank += (kb[p] == key);
      int pos = offs[((size_t)b * NBLK + j) * NBK + key] + cnt[key] + rank;
      sidx[(size_t)b * UTOT + pos] = u;
    }
    __syncthreads();
    if (key >= 0) atomicAdd(&cnt[key], 1);
    __syncthreads();
  }
}

// ---------------- per-batch gathered row vectors (faithful (indices%2)*L semantics) ----------------
__global__ void prep_kernel(const float* __restrict__ f1f2, const float* __restrict__ refw,
                            const int* __restrict__ rindex, float* __restrict__ Pbuf) {
  int tid = threadIdx.x;            // 128 threads
  if (tid >= Bn * 64) return;
  int b = tid / 64, e = tid % 64;
  int ri0 = rindex[0], riL = rindex[Ln];
  float* Pb = Pbuf + b * 256;
  if (e < 16) {
    float q0 = f1f2[((size_t)b * L2n + ri0) * Cmn + e];
    float q1 = f1f2[((size_t)b * L2n + riL) * Cmn + e];
    Pb[e] = q0; Pb[16 + e] = q1;
    float n0 = 0.0f, n1 = 0.0f;
    for (int k = 0; k < 16; ++k) {
      float a0 = f1f2[((size_t)b * L2n + ri0) * Cmn + k];
      float a1 = f1f2[((size_t)b * L2n + riL) * Cmn + k];
      n0 += a0 * a0; n1 += a1 * a1;
    }
    n0 = fmaxf(sqrtf(n0), 5e-5f); n1 = fmaxf(sqrtf(n1), 5e-5f);
    Pb[32 + e] = q0 / n0; Pb[48 + e] = q1 / n1;
  }
  Pb[64 + e]  = refw[((size_t)b * L2n + ri0) * C0n + e];
  Pb[128 + e] = refw[((size_t)b * L2n + riL) * C0n + e];
  if (e == 0) {
    Pb[192] = (ri0 < Ln) ? 0.01f : 0.99f;
    Pb[193] = (riL < Ln) ? 0.01f : 0.99f;
  }
}

// ---------------- chunked LSH attention: WMMA gemm1 -> exp softmax -> WMMA gemm2 ----------------
__global__ __launch_bounds__(288) void att_kernel(const float* __restrict__ Pbuf,
                                                  const int* __restrict__ sidx,
                                                  float* __restrict__ bs_un,
                                                  float* __restrict__ ret_un) {
  extern __shared__ char smem[];
  __bf16* qA   = (__bf16*)smem;                 // 144 x 32
  __bf16* kmB  = qA  + 144 * 32;                // 432 x 32
  __bf16* Wl   = kmB + 432 * 32;                // 144 x 448
  __bf16* rbT  = Wl  + 144 * WSTR;              // 64  x 448
  float*  zl   = (float*)(rbT + 64 * WSTR);     // 432
  float*  denomL = zl + 432;                    // 144
  int*    rowdst = (int*)(denomL + 144);        // 144

  int bid = blockIdx.x;
  int kc = bid % NCHK;
  int h  = (bid / NCHK) % NHASH;
  int b  = bid / (NCHK * NHASH);
  const float* Pb = Pbuf + b * 256;
  const int* idxb = sidx + (size_t)b * UTOT;
  int tid = threadIdx.x;

  // ---- fill q rows (i dimension) ----
  if (tid < CHK) {
    int i = tid;
    int s = kc * CHK + i;
    int sp = (s < L2n) ? s : (s - 16);          // padding duplicated tail rows
    int u = idxb[h * L2n + sp];
    int par = u & 1;
#pragma unroll
    for (int e = 0; e < 16; ++e) qA[i * 32 + e] = (__bf16)Pb[par * 16 + e];
#pragma unroll
    for (int e = 16; e < 32; ++e) qA[i * 32 + e] = (__bf16)0.0f;
    rowdst[i] = (s < L2n) ? u : -1;
  }
  // ---- fill km / z / rbT (j dimension: [self, roll+1, roll-1] chunks) ----
  for (int j = tid; j < 432; j += 288) {
    int d = j / CHK, jj = j % CHK;
    int c = kc;
    if (d == 1) c = (kc + NCHK - 1) % NCHK;
    else if (d == 2) c = (kc + 1) % NCHK;
    int s = c * CHK + jj;
    int sp = (s < L2n) ? s : (s - 16);
    int u = idxb[h * L2n + sp];
    int par = u & 1;
#pragma unroll
    for (int e = 0; e < 16; ++e) kmB[j * 32 + e] = (__bf16)Pb[32 + par * 16 + e];
#pragma unroll
    for (int e = 16; e < 32; ++e) kmB[j * 32 + e] = (__bf16)0.0f;
    zl[j] = Pb[192 + par];
    for (int e = 0; e < 64; ++e) rbT[e * WSTR + j] = (__bf16)Pb[64 + par * 64 + e];
  }
  // ---- zero K padding (cols 432..447) ----
  for (int x = tid; x < 144 * 16; x += 288) Wl[(x / 16) * WSTR + 432 + (x % 16)] = (__bf16)0.0f;
  for (int x = tid; x < 64 * 16; x += 288)  rbT[(x / 16) * WSTR + 432 + (x % 16)] = (__bf16)0.0f;
  __syncthreads();

  int wv = tid >> 5, lane = tid & 31;
  int ln = lane & 15, half = lane >> 4;

  // ---- gemm1: raw = qb (144x16) * km^T -> exp * z; per-lane column partial sums ----
  {
    int it = wv;                                // 9 waves, one per 16-row block
    float racc[8];
#pragma unroll
    for (int v = 0; v < 8; ++v) racc[v] = 0.0f;
    v16bf a;
    const __bf16* ar = &qA[(it * 16 + ln) * 32];
#pragma unroll
    for (int jx = 0; jx < 16; ++jx) a[jx] = ar[16 * half + jx];
    for (int jt = 0; jt < 27; ++jt) {
      v16bf bb;
      const __bf16* br = &kmB[(jt * 16 + ln) * 32];
#pragma unroll
      for (int jx = 0; jx < 16; ++jx) bb[jx] = br[16 * half + jx];
      v8f c = {};
      c = __builtin_amdgcn_wmma_f32_16x16x32_bf16(false, a, false, bb, (short)0, c, false, false);
      int col = jt * 16 + ln;
      float zc = zl[col];
#pragma unroll
      for (int v = 0; v < 8; ++v) {
        float wexp = __expf(c[v]) * zc;
        int row = it * 16 + v + 8 * half;
        Wl[row * WSTR + col] = (__bf16)wexp;
        racc[v] += wexp;                        // this lane's columns only
      }
    }
    // single cross-lane reduction after the K loop (deterministic tree)
#pragma unroll
    for (int v = 0; v < 8; ++v) {
      float rs = racc[v];
      rs += __shfl_xor(rs, 1, 32);
      rs += __shfl_xor(rs, 2, 32);
      rs += __shfl_xor(rs, 4, 32);
      rs += __shfl_xor(rs, 8, 32);
      racc[v] = rs;
    }
    if (ln == 0) {
#pragma unroll
      for (int v = 0; v < 8; ++v) denomL[it * 16 + v + 8 * half] = racc[v];
    }
  }
  __syncthreads();

  // ---- bucket_score scatter into unsorted (h_orig, t) layout ----
  if (tid < CHK) {
    int u = rowdst[tid];
    if (u >= 0) {
      int ho = u / L2n, t = u % L2n;
      bs_un[((size_t)b * NHASH + ho) * L2n + t] = denomL[tid];
    }
  }

  // ---- gemm2: ret = (W * rb) / denom, scattered to unsorted layout ----
  {
    int it = wv;
    const __bf16* wrow = &Wl[(it * 16 + ln) * WSTR];
    for (int et = 0; et < 4; ++et) {
      const __bf16* brow = &rbT[(et * 16 + ln) * WSTR];
      v8f acc = {};
      for (int kt = 0; kt < 14; ++kt) {
        v16bf a, bb;
        int ka = kt * 32 + 16 * half;
#pragma unroll
        for (int jx = 0; jx < 16; ++jx) a[jx] = wrow[ka + jx];
        int kb = kt * 32 + 16 * half;
#pragma unroll
        for (int jx = 0; jx < 16; ++jx) bb[jx] = brow[kb + jx];
        acc = __builtin_amdgcn_wmma_f32_16x16x32_bf16(false, a, false, bb, (short)0, acc, false, false);
      }
      int e = et * 16 + ln;
#pragma unroll
      for (int v = 0; v < 8; ++v) {
        int row = it * 16 + v + 8 * half;
        int u = rowdst[row];
        if (u >= 0) {
          int ho = u / L2n, t = u % L2n;
          ret_un[(((size_t)b * NHASH + ho) * L2n + t) * 64 + e] = acc[v] / denomL[row];
        }
      }
    }
  }
}

// ---------------- final: prob-weighted mix over hash rounds + inverse permutation ----------------
__global__ void final_kernel(const float* __restrict__ ret_un, const float* __restrict__ bs_un,
                             const int* __restrict__ rindex, float* __restrict__ out) {
  int id = blockIdx.x * blockDim.x + threadIdx.x;
  if (id >= Bn * L2n * 64) return;
  int c = id % 64;
  int t = (id / 64) % L2n;
  int b = id / (64 * L2n);
  float bsv[NHASH], bsum = 0.0f;
#pragma unroll
  for (int ho = 0; ho < NHASH; ++ho) {
    bsv[ho] = bs_un[((size_t)b * NHASH + ho) * L2n + t];
    bsum += bsv[ho];
  }
  int p = rindex[t];
  if (p < Ln) {
    float acc = 0.0f;
#pragma unroll
    for (int ho = 0; ho < NHASH; ++ho)
      acc += ret_un[(((size_t)b * NHASH + ho) * L2n + t) * 64 + c] * (bsv[ho] / bsum);
    out[((size_t)(b * 64 + c)) * Ln + p] = acc;
  }
}

// ---------------- host launcher ----------------
extern "C" void kernel_launch(void* const* d_in, const int* in_sizes, int n_in,
                              void* d_out, int out_size, void* d_ws, size_t ws_size,
                              hipStream_t stream) {
  (void)in_sizes; (void)n_in; (void)out_size; (void)ws_size;
  const float* fd1 = (const float*)d_in[0];
  const float* fd2 = (const float*)d_in[1];
  const float* ref = (const float*)d_in[2];
  const int* rindex = (const int*)d_in[3];
  const float* mw1 = (const float*)d_in[4];  const float* mb1 = (const float*)d_in[5];
  const float* mw2 = (const float*)d_in[6];  const float* mb2 = (const float*)d_in[7];
  const float* mws = (const float*)d_in[8];  const float* mbs = (const float*)d_in[9];
  const float* a1w1 = (const float*)d_in[10]; const float* a1b1 = (const float*)d_in[11];
  const float* a1w2 = (const float*)d_in[12]; const float* a1b2 = (const float*)d_in[13];
  const float* a2w1 = (const float*)d_in[14]; const float* a2b1 = (const float*)d_in[15];
  const float* a2w2 = (const float*)d_in[16]; const float* a2b2 = (const float*)d_in[17];
  float* out = (float*)d_out;

  char* ws = (char*)d_ws;
  size_t off = 0;
  auto alloc = [&](size_t bytes) -> char* {
    char* p = ws + off;
    off = (off + bytes + 255) & ~(size_t)255;
    return p;
  };
  float* rot   = (float*)alloc((size_t)16 * NHASH * 64 * 4);
  float* f1f2  = (float*)alloc((size_t)Bn * L2n * Cmn * 4);
  float* refw  = (float*)alloc((size_t)Bn * L2n * C0n * 4);
  float* tmp16 = (float*)alloc((size_t)Bn * Cmn * Ln * 4);
  float* tmp64 = (float*)alloc((size_t)Bn * C0n * Ln * 4);
  int*   ckey  = (int*)  alloc((size_t)Bn * UTOT * 4);
  int*   hist  = (int*)  alloc((size_t)Bn * NBLK * NBK * 4);
  int*   offsb = (int*)  alloc((size_t)Bn * NBLK * NBK * 4);
  int*   sidx  = (int*)  alloc((size_t)Bn * UTOT * 4);
  float* bs_un = (float*)alloc((size_t)Bn * NHASH * L2n * 4);
  float* retun = (float*)alloc((size_t)Bn * NHASH * L2n * 64 * 4);
  float* Pbuf  = (float*)alloc((size_t)Bn * 256 * 4);

  hipFuncSetAttribute(reinterpret_cast<const void*>(att_kernel),
                      hipFuncAttributeMaxDynamicSharedMemorySize, SMEM_ATT);

  const int T = 256;
  // convs: m-branch (f1 then f2) and aux branches (aux1 then refe)
  int nm = Bn * Cmn * Ln, na = Bn * C0n * Ln;
  conv3x3_kernel<<<(na + T - 1) / T, T, 0, stream>>>(fd1, mw1, mb1, tmp16, C0n, Cmn, 1);
  m_stage2_kernel<<<(nm + T - 1) / T, T, 0, stream>>>(fd1, tmp16, mw2, mb2, mws, mbs, f1f2, 0);
  conv3x3_kernel<<<(na + T - 1) / T, T, 0, stream>>>(fd2, mw1, mb1, tmp16, C0n, Cmn, 1);
  m_stage2_kernel<<<(nm + T - 1) / T, T, 0, stream>>>(fd2, tmp16, mw2, mb2, mws, mbs, f1f2, Ln);
  conv3x3_kernel<<<(na + T - 1) / T, T, 0, stream>>>(fd1, a1w1, a1b1, tmp64, C0n, C0n, 1);
  aux_stage2_kernel<<<(na + T - 1) / T, T, 0, stream>>>(fd1, tmp64, a1w2, a1b2, refw, 0);
  conv3x3_kernel<<<(na + T - 1) / T, T, 0, stream>>>(ref, a2w1, a2b1, tmp64, C0n, C0n, 1);
  aux_stage2_kernel<<<(na + T - 1) / T, T, 0, stream>>>(ref, tmp64, a2w2, a2b2, refw, Ln);

  rot_kernel<<<16, 256, 0, stream>>>(rot);
  codes_kernel<<<(Bn * UTOT + T - 1) / T, T, 0, stream>>>(f1f2, rot, rindex, ckey);

  hist_kernel<<<Bn * NBLK, 256, 0, stream>>>(ckey, hist);
  scan_kernel<<<Bn, NBK, 0, stream>>>(hist, offsb);
  scatter_kernel<<<Bn * NBLK, 256, 0, stream>>>(ckey, offsb, sidx);

  prep_kernel<<<1, 128, 0, stream>>>(f1f2, refw, rindex, Pbuf);

  att_kernel<<<Bn * NHASH * NCHK, 288, SMEM_ATT, stream>>>(Pbuf, sidx, bs_un, retun);

  final_kernel<<<(Bn * L2n * 64 + T - 1) / T, T, 0, stream>>>(retun, bs_un, rindex, out);
}
